// Net_30571577213510
// MI455X (gfx1250) — matrix operand order repaired and checked by reading, not compiled, not run on hardware
//
#include <hip/hip_runtime.h>

typedef __attribute__((ext_vector_type(16))) _Float16 v16h;
typedef __attribute__((ext_vector_type(8)))  float    v8f;
typedef unsigned long long ull;
typedef union { ull d[4]; v16h h; } BF;

#define BETA      0.9f
#define THRESH    1.0f
#define NSTEP     100
#define H         256
#define BATCH     4096
#define NT        8            // row tiles per WG (one per wave)
#define KB        8            // 256 / 32 k-blocks
#define ROWS_PER_WG 128
// A-fragment LDS: per fragment 32 lanes * 10 dwords (8 payload + 2 pad).
// Lane stride 10 dwords: banks (10l+c) mod 64, gcd(5,32)=1 -> every wave32
// ds_load_b64/store_b64 hits all 64 banks exactly once (conflict-free).
#define FRAG_DW   320          // dwords per fragment slot
#define ABUF_DW   (NT * KB * FRAG_DW)   // 20480 dwords = 80 KB per buffer
#define SCRATCH_OFF (2 * ABUF_DW)       // reduction scratch (floats)
#define LDS_DW    (2 * ABUF_DW + NT * ROWS_PER_WG)  // 41984 dw = 164 KB

extern "C" __global__
__attribute__((amdgpu_waves_per_eu(1)))   // weight-stationary: ~700 VGPRs/wave
__launch_bounds__(256)
void snn_lif_wmma_kernel(const float* __restrict__ x,
                         const float* __restrict__ W1,
                         const float* __restrict__ b1,
                         const float* __restrict__ W2,
                         const float* __restrict__ b2,
                         const float* __restrict__ W3,
                         const float* __restrict__ b3,
                         float* __restrict__ out)
{
    extern __shared__ unsigned int lds[];

    const int tid   = threadIdx.x;
    const int lane  = tid & 31;
    const int wav   = tid >> 5;      // owns row tile 'wav' and cols [32w,32w+32)
    const int half  = lane >> 4;
    const int lan16 = lane & 15;
    const int wgRow = blockIdx.x * ROWS_PER_WG;

    // ---------------------------------------------------------------------
    // Weight-stationary B: this wave holds W2 columns nb in {2w, 2w+1} for
    // all 8 k-blocks, as split-f16 (hi = f16(w), lo = f16(w - hi)) so
    // spk1@hi + spk1@lo reproduces fp32 W2 to ~2^-22 (spikes are exact 0/1).
    // B layout: lane l = column N = nb*16+(l&15); VGPR r packs
    // (K = kb*32 + (l>=16?8:0) + (r<4 ? 2r : 16+2(r-4)), +1).   32 x v16h = 256 VGPRs.
    // ---------------------------------------------------------------------
    v16h Bh[2][KB], Bl[2][KB];
    #pragma unroll
    for (int j = 0; j < 2; ++j) {
        const int n = (2 * wav + j) * 16 + lan16;
        #pragma unroll
        for (int kb = 0; kb < KB; ++kb) {
            #pragma unroll
            for (int r = 0; r < 8; ++r) {
                #pragma unroll
                for (int hh = 0; hh < 2; ++hh) {
                    int k = kb * 32 + (half ? 8 : 0) + ((r < 4) ? (2 * r + hh) : (16 + 2 * (r - 4) + hh));
                    float w = W2[k * H + n];
                    _Float16 whi = (_Float16)w;
                    _Float16 wlo = (_Float16)(w - (float)whi);
                    Bh[j][kb][2 * r + hh] = whi;
                    Bl[j][kb][2 * r + hh] = wlo;
                }
            }
        }
    }

    // ---------------------------------------------------------------------
    // Layer-1 state for this wave's own 16 rows, in A-fragment layout:
    //   idx = kb*16 + 2r + h  <->  row = lane&15,
    //   col = kb*32 + (r<4 ? 2r+h : 16+2(r-4)+h) + 8*(lane>>4)
    // ---------------------------------------------------------------------
    const int myRow = wgRow + wav * 16 + lan16;
    const float x0 = x[myRow * 2 + 0];
    const float x1 = x[myRow * 2 + 1];

    float cur1f[128];
    float mem1f[128];
    #pragma unroll
    for (int kb = 0; kb < KB; ++kb) {
        #pragma unroll
        for (int r = 0; r < 8; ++r) {
            #pragma unroll
            for (int h = 0; h < 2; ++h) {
                int kk  = (r < 4) ? (2 * r + h) : (16 + 2 * (r - 4) + h);
                int c   = kb * 32 + kk + half * 8;
                int idx = kb * 16 + 2 * r + h;
                cur1f[idx] = __builtin_fmaf(x0, W1[c],
                             __builtin_fmaf(x1, W1[H + c], b1[c]));
                mem1f[idx] = 0.0f;
            }
        }
    }

    float b2v[2], w3v[2];
    #pragma unroll
    for (int j = 0; j < 2; ++j) {
        b2v[j] = b2[(2 * wav + j) * 16 + lan16];
        w3v[j] = W3[(2 * wav + j) * 16 + lan16];
    }
    const float b3s = b3[0];

    // mem2 == accumulator.  acc[tau*2+j][r] <-> row = wgRow+tau*16+8*half+r,
    // col = (2*wav+j)*16 + lan16.   (all 128 rows x this wave's 32 cols)
    v8f acc[16];
    #pragma unroll
    for (int i = 0; i < 16; ++i)
        acc[i] = (v8f){0.f,0.f,0.f,0.f,0.f,0.f,0.f,0.f};

    // ---------------------------------------------------------------------
    // 100 sequential LIF steps.  Per step: LIF2-fold into acc; LIF1 on own
    // rows -> write 8 A-fragments (spikes) to LDS (double-buffered by t&1);
    // one barrier; then 64 A-fragments x 4 WMMAs against resident B regs.
    // reset(t+1) == spk(t) == (mem > 1), so spikes are never stored.
    // ---------------------------------------------------------------------
    #pragma clang loop unroll(disable)
    for (int t = 0; t < NSTEP; ++t) {
        unsigned int* bufb = lds + (t & 1) * ABUF_DW + lane * 10;

        // acc = BETA*mem2 + b2 - (mem2 > 1); GEMM then adds cur2 - b2.
        #pragma unroll
        for (int i = 0; i < 16; ++i) {
            v8f m = acc[i];
            v8f a;
            #pragma unroll
            for (int r = 0; r < 8; ++r) {
                float mm  = m[r];
                float rst = (mm > THRESH) ? 1.0f : 0.0f;
                a[r] = __builtin_fmaf(BETA, mm, b2v[i & 1] - rst);
            }
            acc[i] = a;
        }

        // LIF1 + spike A-fragments for own tile -> LDS.
        {
            unsigned int* wb = bufb + wav * (KB * FRAG_DW);
            #pragma unroll
            for (int kb = 0; kb < KB; ++kb) {
                BF af;
                #pragma unroll
                for (int j = 0; j < 16; ++j) {
                    int   idx = kb * 16 + j;
                    float m   = mem1f[idx];
                    float rst = (m > THRESH) ? 1.0f : 0.0f;
                    m = __builtin_fmaf(BETA, m, cur1f[idx] - rst);
                    mem1f[idx] = m;
                    af.h[j] = (m > THRESH) ? (_Float16)1.0f : (_Float16)0.0f;
                }
                ull* qs = (ull*)(wb + kb * FRAG_DW);
                qs[0] = af.d[0]; qs[1] = af.d[1];
                qs[2] = af.d[2]; qs[3] = af.d[3];
            }
        }
        __syncthreads();   // writes(t) visible before reads(t); t-parity
                           // double buffering makes one barrier sufficient

        // GEMM: 64 A-fragments, each feeding 4 WMMAs on resident B.
        // g -> (tau = g&7, kb = g>>3): consecutive groups rotate through all
        // 8 tau (16 accumulators) so a given acc is revisited only every 8
        // groups -> no WMMA->WMMA accumulation chains.  4-slot A ring with
        // prefetch distance 2: the slot a load overwrites was last read 8
        // WMMAs ago -> outside the WMMA WAR-hazard window (no v_nops), and
        // s_wait_dscnt keeps 2 groups of loads in flight.
        const ull* tb[NT];
        #pragma unroll
        for (int tau = 0; tau < NT; ++tau)
            tb[tau] = (const ull*)bufb + tau * (KB * FRAG_DW / 2);

        BF ab[4];
        #pragma unroll
        for (int p0 = 0; p0 < 2; ++p0) {
            const ull* q = tb[p0 & 7];   // g = 0,1 -> tau = g, kb = 0
            ab[p0].d[0] = q[0]; ab[p0].d[1] = q[1];
            ab[p0].d[2] = q[2]; ab[p0].d[3] = q[3];
        }

        #pragma unroll
        for (int g = 0; g < NT * KB; ++g) {
            if (g + 2 < NT * KB) {
                const int gp = g + 2;
                const ull* q = tb[gp & 7] + (gp >> 3) * (FRAG_DW / 2);
                BF& nx = ab[gp & 3];
                nx.d[0] = q[0]; nx.d[1] = q[1];
                nx.d[2] = q[2]; nx.d[3] = q[3];
            }
            const int tau = g & 7, kb = g >> 3;
            const v16h a = ab[g & 3].h;
            acc[tau*2+0] = __builtin_amdgcn_wmma_f32_16x16x32_f16(
                false, a, false, Bh[0][kb], (short)0, acc[tau*2+0], false, false);
            acc[tau*2+1] = __builtin_amdgcn_wmma_f32_16x16x32_f16(
                false, a, false, Bh[1][kb], (short)0, acc[tau*2+1], false, false);
            acc[tau*2+0] = __builtin_amdgcn_wmma_f32_16x16x32_f16(
                false, a, false, Bl[0][kb], (short)0, acc[tau*2+0], false, false);
            acc[tau*2+1] = __builtin_amdgcn_wmma_f32_16x16x32_f16(
                false, a, false, Bl[1][kb], (short)0, acc[tau*2+1], false, false);
#if __has_builtin(__builtin_amdgcn_sched_group_barrier)
            __builtin_amdgcn_sched_group_barrier(0x008, 4, 0); // 4 WMMA (g)
            __builtin_amdgcn_sched_group_barrier(0x100, 4, 0); // 4 DS rd (g+2)
#endif
        }
    }

    // ---------------------------------------------------------------------
    // Output: cur3 = spk2(final) @ W3 + b3.  Each wave reduces its 32 cols
    // per row (shfl_xor over the 16-lane column group), parks per-wave row
    // partials in LDS, then 128 threads sum the 8 waves.
    // ---------------------------------------------------------------------
    float* fs = (float*)(lds + SCRATCH_OFF);
    #pragma unroll
    for (int tau = 0; tau < NT; ++tau) {
        float p[8];
        #pragma unroll
        for (int r = 0; r < 8; ++r) {
            float s0 = (acc[tau*2+0][r] > THRESH) ? 1.0f : 0.0f;
            float s1 = (acc[tau*2+1][r] > THRESH) ? 1.0f : 0.0f;
            p[r] = __builtin_fmaf(s0, w3v[0], s1 * w3v[1]);
        }
        #pragma unroll
        for (int r = 0; r < 8; ++r) {
            #pragma unroll
            for (int off = 1; off < 16; off <<= 1)
                p[r] += __shfl_xor(p[r], off, 32);
        }
        #pragma unroll
        for (int r = 0; r < 8; ++r) {
            if (lan16 == r)
                fs[wav * ROWS_PER_WG + tau * 16 + half * 8 + r] = p[r];
        }
    }
    __syncthreads();
    if (tid < ROWS_PER_WG) {
        float s = 0.0f;
        #pragma unroll
        for (int w = 0; w < NT; ++w)
            s += fs[w * ROWS_PER_WG + tid];
        out[wgRow + tid] = s + b3s;
    }
}

extern "C" void kernel_launch(void* const* d_in, const int* in_sizes, int n_in,
                              void* d_out, int out_size, void* d_ws, size_t ws_size,
                              hipStream_t stream) {
    (void)in_sizes; (void)n_in; (void)out_size; (void)d_ws; (void)ws_size;
    const float* x  = (const float*)d_in[0];
    const float* W1 = (const float*)d_in[1];
    const float* b1 = (const float*)d_in[2];
    const float* W2 = (const float*)d_in[3];
    const float* b2 = (const float*)d_in[4];
    const float* W3 = (const float*)d_in[5];
    const float* b3 = (const float*)d_in[6];
    float* out = (float*)d_out;

    dim3 grid(BATCH / ROWS_PER_WG);   // 32 workgroups
    dim3 block(256);                  // 8 wave32 waves
    size_t ldsBytes = (size_t)LDS_DW * sizeof(unsigned int); // 164 KB
    hipLaunchKernelGGL(snn_lif_wmma_kernel, grid, block, ldsBytes, stream,
                       x, W1, b1, W2, b2, W3, b3, out);
}
